// Attention_24498493456696
// MI455X (gfx1250) — matrix operand order repaired and checked by reading, not compiled, not run on hardware
//
#include <hip/hip_runtime.h>
#include <hip/hip_bf16.h>
#include <stdint.h>

// ---------------------------------------------------------------------------
// Problem constants (match the reference)
// ---------------------------------------------------------------------------
#define B_  8
#define N_  1024
#define C_  1024
#define H_  16
#define D_  64
#define M_  (B_ * N_)   // 8192 tokens total
#define F_  (3 * C_)    // 3072 qkv features

typedef __attribute__((ext_vector_type(16))) _Float16 v16h;
typedef __attribute__((ext_vector_type(8)))  float    v8f;

union FragH { v16h v; uint32_t u[8]; };
union FragF { v8f  v; float    f[8]; };

__device__ __forceinline__ v8f vzero8() {
    v8f z = {0.f, 0.f, 0.f, 0.f, 0.f, 0.f, 0.f, 0.f};
    return z;
}

// ---------------------------------------------------------------------------
// CDNA5 async global->LDS copy (ASYNCcnt-tracked, bypasses VGPRs).
// LDS flat addresses carry the wave-relative LDS offset in the low 32 bits
// (ISA 10.2 aperture rules), so truncating the generic pointer is the VDST.
// ---------------------------------------------------------------------------
__device__ __forceinline__ void async_b128(_Float16* lds, const _Float16* g) {
    const uint32_t l = (uint32_t)(size_t)lds;
    asm volatile("global_load_async_to_lds_b128 %0, %1, off"
                 :: "v"(l), "v"(g) : "memory");
}
__device__ __forceinline__ void wait_async0() {
    asm volatile("s_wait_asynccnt 0x0" ::: "memory");
}

// ---------------------------------------------------------------------------
// WMMA fragment loaders (layouts per CDNA5 ISA 7.12.2, wave32).
// A matrix 16x32 f16: lane L holds row r=L&15; group g=L>>4;
//   vgpr v holds K = (v<4?0:16) + 8*g + 2*(v&3) and K+1 (contiguous pair).
// B matrix 32x16 f16 stored as [n][k] (k contiguous): lane L holds col n=L&15;
//   vgpr v holds K = (v<4?0:16) + 8*g + 2*(v&3) and K+1 (same as A).
// C matrix 16x16 f32: vgpr v holds (row = v + 8*g, col = L&15).
// ---------------------------------------------------------------------------
__device__ __forceinline__ v16h load_frag_a(const _Float16* __restrict__ base,
                                            int ldh, int kBase, int lane) {
    const int r = lane & 15, g = lane >> 4;
    FragH fr;
    const int rowoff = r * ldh + kBase;
#pragma unroll
    for (int v = 0; v < 8; ++v) {
        const int k = ((v & 4) ? 16 : 0) + g * 8 + 2 * (v & 3);
        fr.u[v] = *(const uint32_t*)(base + rowoff + k);
    }
    return fr.v;
}

__device__ __forceinline__ v16h load_frag_b(const _Float16* __restrict__ base,
                                            int ldh, int kBase, int lane) {
    const int n = lane & 15, g = lane >> 4;
    FragH fr;
    const int rowoff = n * ldh + kBase;
#pragma unroll
    for (int v = 0; v < 8; ++v)
        fr.u[v] = *(const uint32_t*)(base + rowoff + ((v & 4) ? 16 : 0) + g * 8 + 2 * (v & 3));
    return fr.v;
}

// ---------------------------------------------------------------------------
// Kernel 1: f32 -> f16 conversion (vectorized by pairs)
// ---------------------------------------------------------------------------
__global__ void cvt_f2h(const float2* __restrict__ s, uint32_t* __restrict__ d, int n2) {
    int i = blockIdx.x * blockDim.x + threadIdx.x;
    if (i < n2) {
        float2 t = s[i];
        union { uint32_t u; _Float16 h[2]; } cv;
        cv.h[0] = (_Float16)t.x;
        cv.h[1] = (_Float16)t.y;
        d[i] = cv.u;
    }
}

// ---------------------------------------------------------------------------
// Kernel 2/5: tiled WMMA GEMM.  C[m, n] = sum_k A[m,k] * Bw[n,k]
// Tile: BM=64, BN=128, BK=32.  128 threads = 4 waves; wave w owns 16 M-rows
// and all 128 N-cols (1 A-frag feeds 8 WMMAs per K-chunk).
// Global->LDS staging is double-buffered via async-to-LDS b128 copies:
//   issue chunk c+1, compute chunk c, s_wait_asynccnt 0, barrier.
// LDS row stride 40 halfs = 80 B: 16B-aligned for b128, and 20-dword row
// stride keeps the 16-row fragment gathers bank-conflict-free (gcd(5,16)=1).
// MODE 0: scatter f16 qkv results into [B,H,N,D] q/k/v buffers.
// MODE 1: f32 output + bias (final projection).
// ---------------------------------------------------------------------------
template <int MODE>
__global__ __launch_bounds__(128) void gemm_wmma(
    const _Float16* __restrict__ A,     // [M x K]
    const _Float16* __restrict__ Bw,    // [Nn x K]
    int K, int Nn,
    float* __restrict__ outF, const float* __restrict__ bias,        // MODE 1
    _Float16* __restrict__ qh, _Float16* __restrict__ kh,            // MODE 0
    _Float16* __restrict__ vh)
{
    constexpr int LDK = 40;
    __shared__ alignas(16) _Float16 As[2][64 * LDK];
    __shared__ alignas(16) _Float16 Bs[2][128 * LDK];

    const int tid  = threadIdx.x;
    const int lane = tid & 31;
    const int wave = tid >> 5;
    const int m0 = blockIdx.y * 64;
    const int n0 = blockIdx.x * 128;

    v8f acc[8];
#pragma unroll
    for (int n = 0; n < 8; ++n) acc[n] = vzero8();

    // A tile staging: 2 threads per row, 32 B (2 x b128) each
    const int arow = tid >> 1, apart = tid & 1;
    const _Float16* aSrc = A + (size_t)(m0 + arow) * K + apart * 16;
    const int aOff = arow * LDK + apart * 16;
    // B tile staging: 1 thread per row, 64 B (4 x b128) each
    const _Float16* bSrc = Bw + (size_t)(n0 + tid) * K;
    const int bOff = tid * LDK;

    auto stage = [&](int kt, int buf) {
        async_b128(&As[buf][aOff],      aSrc + kt);
        async_b128(&As[buf][aOff + 8],  aSrc + kt + 8);
        async_b128(&Bs[buf][bOff],      bSrc + kt);
        async_b128(&Bs[buf][bOff + 8],  bSrc + kt + 8);
        async_b128(&Bs[buf][bOff + 16], bSrc + kt + 16);
        async_b128(&Bs[buf][bOff + 24], bSrc + kt + 24);
    };

    stage(0, 0);
    wait_async0();
    __syncthreads();

    const int nChunks = K >> 5;
    for (int c = 0; c < nChunks; ++c) {
        const int cur = c & 1;
        if (c + 1 < nChunks) stage((c + 1) << 5, cur ^ 1);   // overlap with WMMAs

        const v16h afrag = load_frag_a(&As[cur][wave * 16 * LDK], LDK, 0, lane);
#pragma unroll
        for (int n = 0; n < 8; ++n) {
            const v16h bfrag = load_frag_b(&Bs[cur][n * 16 * LDK], LDK, 0, lane);
            acc[n] = __builtin_amdgcn_wmma_f32_16x16x32_f16(
                false, afrag, false, bfrag, (short)0, acc[n], false, false);
        }
        wait_async0();
        __syncthreads();
    }

    // epilogue
    const int r = lane & 15, g = lane >> 4;
#pragma unroll
    for (int n = 0; n < 8; ++n) {
        FragF c; c.v = acc[n];
#pragma unroll
        for (int v = 0; v < 8; ++v) {
            const int mrow = m0 + wave * 16 + v + 8 * g;
            const int col  = n0 + n * 16 + r;
            if (MODE == 1) {
                outF[(size_t)mrow * Nn + col] = c.f[v] + bias[col];
            } else {
                // col (f index) -> (part, head, d); mrow -> (batch, token)
                const int p = col >> 10, h = (col >> 6) & 15, d = col & 63;
                const int b = mrow >> 10, tok = mrow & 1023;
                _Float16* dst = (p == 0) ? qh : (p == 1 ? kh : vh);
                dst[((size_t)((b * H_ + h) * N_ + tok)) * D_ + d] = (_Float16)c.f[v];
            }
        }
    }
}

// ---------------------------------------------------------------------------
// Kernel 3: in-place LayerNorm over D=64 for q (with 1/sqrt(D) scale) and k.
// One wave (32 lanes) per row; 2 elems per lane; xor-shuffle reductions.
// ---------------------------------------------------------------------------
__global__ __launch_bounds__(256) void qk_layernorm(
    _Float16* __restrict__ qh, _Float16* __restrict__ kh,
    const float* __restrict__ qw, const float* __restrict__ qb,
    const float* __restrict__ kw, const float* __restrict__ kb)
{
    const int rowIdx = blockIdx.x * 8 + (threadIdx.x >> 5);
    const int lane = threadIdx.x & 31;
    const int d0 = lane * 2;
    union PK { uint32_t u; _Float16 h[2]; };

#pragma unroll
    for (int which = 0; which < 2; ++which) {
        _Float16* buf = which ? kh : qh;
        const float* w = which ? kw : qw;
        const float* bb = which ? kb : qb;
        const float postScale = which ? 1.0f : 0.125f;   // D^-0.5 = 1/8

        uint32_t* p = (uint32_t*)(buf + (size_t)rowIdx * D_ + d0);
        PK cv; cv.u = *p;
        float x0 = (float)cv.h[0], x1 = (float)cv.h[1];
        float s = x0 + x1, ss = x0 * x0 + x1 * x1;
#pragma unroll
        for (int m = 16; m >= 1; m >>= 1) {
            s  += __shfl_xor(s,  m, 32);
            ss += __shfl_xor(ss, m, 32);
        }
        const float mu  = s * (1.f / 64.f);
        const float var = ss * (1.f / 64.f) - mu * mu;
        const float inv = rsqrtf(var + 1e-5f);
        const float y0 = ((x0 - mu) * inv * w[d0]     + bb[d0])     * postScale;
        const float y1 = ((x1 - mu) * inv * w[d0 + 1] + bb[d0 + 1]) * postScale;
        cv.h[0] = (_Float16)y0; cv.h[1] = (_Float16)y1;
        *p = cv.u;
    }
}

// ---------------------------------------------------------------------------
// Kernel 4: flash attention.  One block (4 waves) per 64 q-rows of one head.
// Each wave owns 16 q rows; K/V chunks of 64 keys staged in LDS.
// ---------------------------------------------------------------------------
__global__ __launch_bounds__(128) void flash_attn(
    const _Float16* __restrict__ qh, const _Float16* __restrict__ kh,
    const _Float16* __restrict__ vh, _Float16* __restrict__ oh)
{
    constexpr int ST = 66;                  // 33 dwords: odd -> conflict-free
    __shared__ _Float16 Kt[64 * ST];        // [key][d]
    __shared__ _Float16 Vt[64 * ST];        // [d][key] (transposed at load)
    __shared__ _Float16 Ps[4 * 16 * ST];    // per-wave P staging [row][key]

    const int tid = threadIdx.x, lane = tid & 31, wave = tid >> 5;
    const int bh = blockIdx.x;              // b*H + h
    const int qblk = blockIdx.y;
    const size_t bhOff = (size_t)bh * N_ * D_;
    const _Float16* qg = qh + bhOff;
    const _Float16* kg = kh + bhOff;
    const _Float16* vg = vh + bhOff;

    const int r = lane & 15, g = lane >> 4;
    const int qrow0 = qblk * 64 + wave * 16;

    // Q resident in registers as two A-fragments (K = d: 0..31, 32..63)
    v16h qa[2];
#pragma unroll
    for (int s = 0; s < 2; ++s) {
        FragH fr;
#pragma unroll
        for (int v = 0; v < 8; ++v) {
            const int k = s * 32 + ((v & 4) ? 16 : 0) + g * 8 + 2 * (v & 3);
            fr.u[v] = *(const uint32_t*)(qg + (size_t)(qrow0 + r) * D_ + k);
        }
        qa[s] = fr.v;
    }

    v8f o[4];
#pragma unroll
    for (int n = 0; n < 4; ++n) o[n] = vzero8();
    float mrow[8], lrow[8];
#pragma unroll
    for (int v = 0; v < 8; ++v) { mrow[v] = -1e30f; lrow[v] = 0.f; }

    const int ldRow = tid >> 1, ldPart = tid & 1;
    _Float16* psBase = &Ps[wave * 16 * ST];

    for (int j = 0; j < N_; j += 64) {
        // --- cooperative staging: K chunk (natural), V chunk (transposed) ---
        {
            const uint32_t* ks32 = (const uint32_t*)(kg + (size_t)(j + ldRow) * D_ + ldPart * 32);
            uint32_t* kd32 = (uint32_t*)&Kt[ldRow * ST + ldPart * 32];
#pragma unroll
            for (int i = 0; i < 16; ++i) kd32[i] = ks32[i];
            const uint32_t* vs32 = (const uint32_t*)(vg + (size_t)(j + ldRow) * D_ + ldPart * 32);
#pragma unroll
            for (int i = 0; i < 16; ++i) {
                union { uint32_t u; _Float16 h[2]; } cv; cv.u = vs32[i];
                const int d0 = ldPart * 32 + 2 * i;
                Vt[d0 * ST + ldRow]       = cv.h[0];
                Vt[(d0 + 1) * ST + ldRow] = cv.h[1];
            }
        }
        __syncthreads();

        // --- S = Q K^T (16 x 64), f32 accumulate ---
        FragF sc[4];
#pragma unroll
        for (int n = 0; n < 4; ++n) {
            v8f accS = vzero8();
#pragma unroll
            for (int s = 0; s < 2; ++s) {
                const v16h bf = load_frag_b(&Kt[n * 16 * ST], ST, s * 32, lane);
                accS = __builtin_amdgcn_wmma_f32_16x16x32_f16(
                    false, qa[s], false, bf, (short)0, accS, false, false);
            }
            sc[n].v = accS;
        }

        // --- online softmax (rows live in 16-lane groups; reduce masks 8..1) ---
#pragma unroll
        for (int v = 0; v < 8; ++v) {
            float mx = fmaxf(fmaxf(sc[0].f[v], sc[1].f[v]),
                             fmaxf(sc[2].f[v], sc[3].f[v]));
#pragma unroll
            for (int msk = 8; msk >= 1; msk >>= 1)
                mx = fmaxf(mx, __shfl_xor(mx, msk, 32));
            const float mnew  = fmaxf(mrow[v], mx);
            const float alpha = __expf(mrow[v] - mnew);
            float rs = 0.f;
#pragma unroll
            for (int n = 0; n < 4; ++n) {
                const float p = __expf(sc[n].f[v] - mnew);
                sc[n].f[v] = p;
                rs += p;
            }
#pragma unroll
            for (int msk = 8; msk >= 1; msk >>= 1)
                rs += __shfl_xor(rs, msk, 32);
            lrow[v] = lrow[v] * alpha + rs;
            mrow[v] = mnew;
            o[0][v] *= alpha; o[1][v] *= alpha; o[2][v] *= alpha; o[3][v] *= alpha;
            // stage P (C layout -> LDS row-major) for A-fragment reload
#pragma unroll
            for (int n = 0; n < 4; ++n)
                psBase[(v + 8 * g) * ST + n * 16 + r] = (_Float16)sc[n].f[v];
        }
        asm volatile("" ::: "memory");   // wave-private LDS: DS ops in-order per wave

        // --- reload P as A-fragments ---
        v16h pa[2];
#pragma unroll
        for (int s = 0; s < 2; ++s) {
            FragH fr;
#pragma unroll
            for (int v = 0; v < 8; ++v) {
                const int k = s * 32 + ((v & 4) ? 16 : 0) + g * 8 + 2 * (v & 3);
                fr.u[v] = *(const uint32_t*)(psBase + r * ST + k);
            }
            pa[s] = fr.v;
        }

        // --- O += P V ---
#pragma unroll
        for (int n = 0; n < 4; ++n) {
#pragma unroll
            for (int s = 0; s < 2; ++s) {
                const v16h bf = load_frag_b(&Vt[n * 16 * ST], ST, s * 32, lane);
                o[n] = __builtin_amdgcn_wmma_f32_16x16x32_f16(
                    false, pa[s], false, bf, (short)0, o[n], false, false);
            }
        }
        __syncthreads();
    }

    // --- epilogue: O / l, scatter into [B, N, C] with c = h*64 + d ---
    const int b = bh >> 4, h = bh & 15;
#pragma unroll
    for (int n = 0; n < 4; ++n) {
        FragF c; c.v = o[n];
#pragma unroll
        for (int v = 0; v < 8; ++v) {
            const int tok = qrow0 + v + 8 * g;
            const int d = n * 16 + r;
            const float val = c.f[v] / lrow[v];
            oh[((size_t)(b * N_ + tok)) * C_ + h * 64 + d] = (_Float16)val;
        }
    }
}

// ---------------------------------------------------------------------------
// Host launcher
// ---------------------------------------------------------------------------
extern "C" void kernel_launch(void* const* d_in, const int* in_sizes, int n_in,
                              void* d_out, int out_size, void* d_ws, size_t ws_size,
                              hipStream_t stream) {
    const float* x      = (const float*)d_in[0];
    const float* w_qkv  = (const float*)d_in[1];
    const float* w_proj = (const float*)d_in[2];
    const float* b_proj = (const float*)d_in[3];
    const float* qnw    = (const float*)d_in[4];
    const float* qnb    = (const float*)d_in[5];
    const float* knw    = (const float*)d_in[6];
    const float* knb    = (const float*)d_in[7];
    float* out = (float*)d_out;

    char* ws = (char*)d_ws;
    size_t off = 0;
    auto take = [&](size_t bytes) {
        char* p = ws + off;
        off += (bytes + 255) & ~(size_t)255;
        return p;
    };
    _Float16* xh     = (_Float16*)take((size_t)M_ * C_ * 2);   // 16.8 MB
    _Float16* wqkvh  = (_Float16*)take((size_t)F_ * C_ * 2);   //  6.3 MB
    _Float16* wprojh = (_Float16*)take((size_t)C_ * C_ * 2);   //  2.1 MB
    _Float16* qh     = (_Float16*)take((size_t)M_ * C_ * 2);   // 16.8 MB
    _Float16* kh     = (_Float16*)take((size_t)M_ * C_ * 2);   // 16.8 MB
    _Float16* vh     = (_Float16*)take((size_t)M_ * C_ * 2);   // 16.8 MB
    _Float16* oh     = (_Float16*)take((size_t)M_ * C_ * 2);   // 16.8 MB
    (void)ws_size; (void)in_sizes; (void)n_in; (void)out_size;

    // 1) precision conversion
    {
        int n2 = M_ * C_ / 2;
        cvt_f2h<<<(n2 + 255) / 256, 256, 0, stream>>>((const float2*)x, (uint32_t*)xh, n2);
        n2 = F_ * C_ / 2;
        cvt_f2h<<<(n2 + 255) / 256, 256, 0, stream>>>((const float2*)w_qkv, (uint32_t*)wqkvh, n2);
        n2 = C_ * C_ / 2;
        cvt_f2h<<<(n2 + 255) / 256, 256, 0, stream>>>((const float2*)w_proj, (uint32_t*)wprojh, n2);
    }

    // 2) QKV projection (scatters into per-head q/k/v layouts)
    gemm_wmma<0><<<dim3(F_ / 128, M_ / 64), 128, 0, stream>>>(
        xh, wqkvh, C_, F_, nullptr, nullptr, qh, kh, vh);

    // 3) QK layernorm (+ q scale) in place
    qk_layernorm<<<(B_ * H_ * N_) / 8, 256, 0, stream>>>(qh, kh, qnw, qnb, knw, knb);

    // 4) flash attention -> oh [B, N, C] f16
    flash_attn<<<dim3(B_ * H_, N_ / 64), 128, 0, stream>>>(qh, kh, vh, oh);

    // 5) output projection with bias -> d_out f32
    gemm_wmma<1><<<dim3(C_ / 128, M_ / 64), 128, 0, stream>>>(
        oh, wprojh, C_, C_, out, b_proj, nullptr, nullptr, nullptr);
}
